// SimpleMoEFromNotebook_67662914781412
// MI455X (gfx1250) — matrix operand order repaired and checked by reading, not compiled, not run on hardware
//
#include <hip/hip_runtime.h>

// ---------------- types ----------------
typedef __attribute__((ext_vector_type(16))) __bf16 v16bf;
typedef __attribute__((ext_vector_type(8)))  __bf16 v8bf;
typedef __attribute__((ext_vector_type(8)))  float  v8f;

// ---------------- problem dims ----------------
static constexpr int DM = 1024;               // D_MODEL
static constexpr int DH = 4096;               // D_HIDDEN
static constexpr int NE = 8;                  // experts
static constexpr int NT = 4096;               // tokens
static constexpr int MT = 64;                 // token rows per workgroup tile
static constexpr int TILES_PER_E = NT / MT;   // 64
static constexpr int WST = 40;                // staged-B row stride in bf16 (80 B, 16B aligned)

// LDS layout (dynamic shared, CDNA5 WGP has 320 KB)
static constexpr unsigned XS_OFF   = 0;                      // 64*1024 bf16 = 131072 B
static constexpr unsigned HC_OFF   = XS_OFF + MT * DM * 2;   // 64*128 bf16  =  16384 B
static constexpr unsigned WS_OFF   = HC_OFF + MT * 128 * 2;  // 128*WST bf16 =  10240 B
static constexpr unsigned SLAB_OFF = WS_OFF + 128 * WST * 2; // 8*16*WST bf16=  10240 B
static constexpr unsigned TOK_OFF  = SLAB_OFF + 8 * 16 * WST * 2;
static constexpr unsigned SMEM_BYTES = TOK_OFF + 256;        // ~168.2 KB total

__device__ __forceinline__ v8f zero8() {
  v8f z;
#pragma unroll
  for (int i = 0; i < 8; ++i) z[i] = 0.0f;
  return z;
}

// Build a 16-element bf16 fragment from two 16-byte LDS chunks.
__device__ __forceinline__ v16bf ld_frag(const __bf16* p0, const __bf16* p1) {
  v8bf lo = *(const v8bf*)p0;
  v8bf hi = *(const v8bf*)p1;
  v16bf r;
#pragma unroll
  for (int i = 0; i < 8; ++i) { r[i] = lo[i]; r[i + 8] = hi[i]; }
  return r;
}

// ---------------- kernel 0: zero counters ----------------
__global__ void moe_init(int* counts, int* cursors) {
  int i = threadIdx.x;
  if (i < NE) { counts[i] = 0; cursors[i] = 0; }
}

// ---------------- kernel 1: gate logits -> argmax expert ----------------
// softmax is monotone, so argmax(softmax(z)) == argmax(z); strict '>' keeps
// the first maximum (matches jnp.argmax tie behavior).
__global__ void moe_gate(const float* __restrict__ x,
                         const float* __restrict__ gw,
                         const float* __restrict__ gb,
                         int* __restrict__ expert,
                         int* __restrict__ counts) {
  int t = blockIdx.x * blockDim.x + threadIdx.x;
  if (t >= NT) return;
  float acc[NE];
#pragma unroll
  for (int e = 0; e < NE; ++e) acc[e] = gb[e];
  const float* xr = x + (size_t)t * DM;
  for (int d = 0; d < DM; ++d) {
    float xv = xr[d];
#pragma unroll
    for (int e = 0; e < NE; ++e) acc[e] += xv * gw[e * DM + d];
  }
  int best = 0;
  float bv = acc[0];
#pragma unroll
  for (int e = 1; e < NE; ++e)
    if (acc[e] > bv) { bv = acc[e]; best = e; }
  expert[t] = best;
  atomicAdd(&counts[best], 1);
}

// ---------------- kernel 2: exclusive prefix over 8 counts ----------------
__global__ void moe_prefix(const int* __restrict__ counts, int* __restrict__ offsets) {
  if (threadIdx.x == 0) {
    int o = 0;
    for (int e = 0; e < NE; ++e) { offsets[e] = o; o += counts[e]; }
  }
}

// ---------------- kernel 3: compact token ids per expert ----------------
__global__ void moe_scatter(const int* __restrict__ expert,
                            const int* __restrict__ offsets,
                            int* __restrict__ cursors,
                            int* __restrict__ list) {
  int t = blockIdx.x * blockDim.x + threadIdx.x;
  if (t >= NT) return;
  int e = expert[t];
  int pos = atomicAdd(&cursors[e], 1);
  list[offsets[e] + pos] = t;
}

// ---------------- kernel 4: fused top-1 MoE MLP (bf16 WMMA) ----------------
__global__ __launch_bounds__(256) void moe_mlp(const float* __restrict__ x,
                                               const float* __restrict__ w1,
                                               const float* __restrict__ b1,
                                               const float* __restrict__ w2,
                                               const float* __restrict__ b2,
                                               const int* __restrict__ counts,
                                               const int* __restrict__ offsets,
                                               const int* __restrict__ list,
                                               float* __restrict__ out) {
  extern __shared__ char smem[];
  __bf16* Xs   = (__bf16*)(smem + XS_OFF);    // [64][1024] bf16 X tile
  __bf16* Hc   = (__bf16*)(smem + HC_OFF);    // [64][128]  bf16 hidden chunk
  __bf16* Ws   = (__bf16*)(smem + WS_OFF);    // [128 n][WST k] staged W1 block
  __bf16* slabsAll = (__bf16*)(smem + SLAB_OFF); // 8 waves x [16 n][WST k] W2 slabs
  int*    tokS = (int*)(smem + TOK_OFF);      // [64] token ids (-1 = pad)

  const int e    = blockIdx.x / TILES_PER_E;
  const int tile = blockIdx.x % TILES_PER_E;
  const int cnt  = counts[e];
  if (tile * MT >= cnt) return;               // uniform across block: safe
  const int base = offsets[e] + tile * MT;

  const int tid  = threadIdx.x;
  const int lane = tid & 31;
  const int wave = tid >> 5;                  // 0..7
  const int half = lane >> 4;                 // 0/1
  const int ln   = lane & 15;                 // 0..15
  const int st   = wave >> 1;                 // phase-A row sub-tile 0..3
  const int ch   = wave & 1;                  // phase-A column half 0/1
  __bf16* slab = slabsAll + wave * 16 * WST;  // wave-private W2 staging

  const size_t w1base = (size_t)e * DM * DH;
  const size_t w2base = (size_t)e * DH * DM;

  if (tid < MT) {
    int r = tile * MT + tid;
    tokS[tid] = (r < cnt) ? list[base + tid] : -1;
  }
  __syncthreads();

  // ---- load X tile (fp32 -> bf16) into LDS; pad rows are zero ----
  for (int i = 0; i < (MT * DM) / 256; ++i) {
    int lin = tid + 256 * i;                  // consecutive d -> coalesced
    int m = lin >> 10;
    int d = lin & (DM - 1);
    int tk = tokS[m];
    Xs[lin] = (__bf16)((tk >= 0) ? x[(size_t)tk * DM + d] : 0.0f);
  }
  __syncthreads();

  // persistent Y accumulators: wave owns output columns [wave*128, wave*128+128)
  v8f accY[4][8];
#pragma unroll
  for (int a = 0; a < 4; ++a)
#pragma unroll
    for (int b = 0; b < 8; ++b) accY[a][b] = zero8();

  for (int hc = 0; hc < DH / 128; ++hc) {
    // ================= phase A: Hc = relu(X @ W1[:, hc*128 : +128] + b1) ====
    v8f accH[4];
#pragma unroll
    for (int jt = 0; jt < 4; ++jt) accH[jt] = zero8();

    for (int s = 0; s < DM / 32; ++s) {
      const int k0 = s * 32;
      // cooperative stage: W1[k0..k0+31, hc*128..+128) -> Ws[n][k] as bf16
#pragma unroll
      for (int i = 0; i < 16; ++i) {
        int lin = tid + 256 * i;
        int kk = lin >> 7;                    // 0..31
        int nn = lin & 127;                   // coalesced along n
        float v = w1[w1base + (size_t)(k0 + kk) * DH + hc * 128 + nn];
        Ws[nn * WST + kk] = (__bf16)v;
      }
      __syncthreads();
      // A fragment: rows st*16..+15, K = k0..k0+31 (CDNA5 16x32 bf16 layout)
      const __bf16* ap = Xs + ((st * 16 + ln) * DM + k0 + half * 8);
      v16bf a = ld_frag(ap, ap + 16);
#pragma unroll
      for (int jt = 0; jt < 4; ++jt) {        // 4 N-tiles reuse staged block
        const __bf16* bp = Ws + ((ch * 64 + jt * 16 + ln) * WST + half * 8);
        v16bf b = ld_frag(bp, bp + 16);
        accH[jt] = __builtin_amdgcn_wmma_f32_16x16x32_bf16(
            false, a, false, b, (short)0, accH[jt], false, false);
      }
      __syncthreads();
    }
    // bias + relu -> Hc (bf16). C/D layout: elem r -> M = r + 8*half, N = ln.
#pragma unroll
    for (int jt = 0; jt < 4; ++jt) {
      int ncol = ch * 64 + jt * 16 + ln;
      float bb = b1[(size_t)e * DH + hc * 128 + ncol];
#pragma unroll
      for (int r = 0; r < 8; ++r) {
        int m = st * 16 + r + half * 8;
        float v = accH[jt][r] + bb;
        Hc[m * 128 + ncol] = (__bf16)(v > 0.0f ? v : 0.0f);
      }
    }
    __syncthreads();

    // ================= phase B: Y += Hc @ W2[hc*128 : +128, :] ==============
    for (int s2 = 0; s2 < 4; ++s2) {
      const int k0 = s2 * 32;
      v16bf af[4];
#pragma unroll
      for (int sub = 0; sub < 4; ++sub) {
        const __bf16* ap = Hc + ((sub * 16 + ln) * 128 + k0 + half * 8);
        af[sub] = ld_frag(ap, ap + 16);
      }
#pragma unroll
      for (int j = 0; j < 8; ++j) {
        // wave-private stage: W2[hc*128+k0..+32, wave*128+j*16..+16) -> slab
#pragma unroll
        for (int i = 0; i < 16; ++i) {
          int lin = lane + 32 * i;
          int kk = lin >> 4;                  // 0..31
          int nn = lin & 15;
          float v = w2[w2base + (size_t)(hc * 128 + k0 + kk) * DM +
                       wave * 128 + j * 16 + nn];
          slab[nn * WST + kk] = (__bf16)v;    // in-wave DS ordering suffices
        }
        const __bf16* bp = slab + (ln * WST + half * 8);
        v16bf b = ld_frag(bp, bp + 16);
#pragma unroll
        for (int sub = 0; sub < 4; ++sub) {   // 4 row sub-tiles reuse B frag
          accY[sub][j] = __builtin_amdgcn_wmma_f32_16x16x32_bf16(
              false, af[sub], false, b, (short)0, accY[sub][j], false, false);
        }
      }
    }
    __syncthreads();                          // Hc reused next hc iteration
  }

  // ---- epilogue: +b2, scatter valid rows to out (fp32) ----
#pragma unroll
  for (int sub = 0; sub < 4; ++sub) {
#pragma unroll
    for (int j = 0; j < 8; ++j) {
      int ncol = wave * 128 + j * 16 + ln;
      float bb = b2[(size_t)e * DM + ncol];
#pragma unroll
      for (int r = 0; r < 8; ++r) {
        int m = sub * 16 + r + half * 8;
        int tk = tokS[m];
        if (tk >= 0) out[(size_t)tk * DM + ncol] = accY[sub][j][r] + bb;
      }
    }
  }
}

// ---------------- host-side launch ----------------
extern "C" void kernel_launch(void* const* d_in, const int* in_sizes, int n_in,
                              void* d_out, int out_size, void* d_ws, size_t ws_size,
                              hipStream_t stream) {
  (void)in_sizes; (void)n_in; (void)out_size; (void)ws_size;
  const float* x  = (const float*)d_in[0];
  const float* gw = (const float*)d_in[1];
  const float* gb = (const float*)d_in[2];
  const float* w1 = (const float*)d_in[3];
  const float* b1 = (const float*)d_in[4];
  const float* w2 = (const float*)d_in[5];
  const float* b2 = (const float*)d_in[6];
  float* out = (float*)d_out;

  // workspace: [0..7] counts, [8..15] offsets, [16..23] cursors,
  // [32..32+4096) expert id per token, [4128..4128+4096) compacted list
  int* wsI     = (int*)d_ws;
  int* counts  = wsI;
  int* offsets = wsI + 8;
  int* cursors = wsI + 16;
  int* expert  = wsI + 32;
  int* list    = wsI + 32 + NT;

  moe_init<<<1, 32, 0, stream>>>(counts, cursors);
  moe_gate<<<NT / 256, 256, 0, stream>>>(x, gw, gb, expert, counts);
  moe_prefix<<<1, 32, 0, stream>>>(counts, offsets);
  moe_scatter<<<NT / 256, 256, 0, stream>>>(expert, offsets, cursors, list);
  moe_mlp<<<NE * TILES_PER_E, 256, SMEM_BYTES, stream>>>(
      x, w1, b1, w2, b2, counts, offsets, list, out);
}
